// Knn_sim_27350351740930
// MI455X (gfx1250) — matrix operand order repaired and checked by reading, not compiled, not run on hardware
//
#include <hip/hip_runtime.h>

// Problem constants (from reference setup_inputs)
#define D      256
#define NB     2048
#define NA     50000
#define BTILE  32          // rows of 'features' per workgroup
#define NSPLIT 4           // anchor-dimension splits
#define SPLIT_LEN (NA / NSPLIT)   // 12500
#define CHUNK  128         // anchors staged in LDS per iteration
#define KTOP   50

typedef __attribute__((ext_vector_type(16))) __bf16        v16bf;
typedef __attribute__((ext_vector_type(8)))  float         v8f;
typedef __attribute__((ext_vector_type(4)))  unsigned int  v4u;
typedef __attribute__((ext_vector_type(4)))  float         v4f;

union FragU { v4u q[2]; v16bf v; };   // 32 bytes: one WMMA A/B fragment per lane

// ---------------- bf16 split helpers (round-to-nearest-even) ----------------
__device__ __forceinline__ unsigned short f2bf(float f) {
  unsigned int u = __float_as_uint(f);
  unsigned int r = ((u >> 16) & 1u) + 0x7fffu;
  return (unsigned short)((u + r) >> 16);
}
__device__ __forceinline__ float bf2f(unsigned short b) {
  return __uint_as_float(((unsigned int)b) << 16);
}

// f32 -> (bf16_hi, bf16_lo) planes; vectorized 4 elems/iter (pure HBM pass).
__global__ void split_convert(const float* __restrict__ src,
                              unsigned short* __restrict__ hi,
                              unsigned short* __restrict__ lo, int n4) {
  for (int i = blockIdx.x * blockDim.x + threadIdx.x; i < n4;
       i += gridDim.x * blockDim.x) {
    v4f x = ((const v4f*)src)[i];
    unsigned short h0 = f2bf(x[0]), h1 = f2bf(x[1]);
    unsigned short h2 = f2bf(x[2]), h3 = f2bf(x[3]);
    unsigned short l0 = f2bf(x[0] - bf2f(h0)), l1 = f2bf(x[1] - bf2f(h1));
    unsigned short l2 = f2bf(x[2] - bf2f(h2)), l3 = f2bf(x[3] - bf2f(h3));
    ((uint2*)hi)[i] = make_uint2((unsigned)h0 | ((unsigned)h1 << 16),
                                 (unsigned)h2 | ((unsigned)h3 << 16));
    ((uint2*)lo)[i] = make_uint2((unsigned)l0 | ((unsigned)l1 << 16),
                                 (unsigned)l2 | ((unsigned)l3 << 16));
  }
}

__device__ __forceinline__ v8f wmma_bf16(v16bf a, v16bf b, v8f c) {
  // D = A(16x32 bf16) * B(32x16 bf16) + C(16x16 f32)
  return __builtin_amdgcn_wmma_f32_16x16x32_bf16(false, a, false, b,
                                                 (short)0, c, false, false);
}

// Stage one 128-anchor chunk (hi+lo planes + labels) global -> LDS, zero-padded.
__device__ __forceinline__ void stage_chunk(const unsigned short* __restrict__ Ahi,
                                            const unsigned short* __restrict__ Alo,
                                            const int* __restrict__ alabels,
                                            unsigned short* sB_hi,
                                            unsigned short* sB_lo,
                                            int* aLab, int splitBase, int cc,
                                            int t0, int tstride) {
  const int cBase  = cc * CHUNK;
  const int rem    = SPLIT_LEN - cBase;
  const int cCount = (rem < CHUNK) ? rem : CHUNK;
  const int valid4 = cCount * (D / 8);
  const int total4 = CHUNK * (D / 8);
  const size_t gbase = (size_t)(splitBase + cBase) * D;
  for (int i = t0; i < total4; i += tstride) {
    v4u vh = {0, 0, 0, 0}, vl = {0, 0, 0, 0};
    if (i < valid4) {
      vh = *(const v4u*)&Ahi[gbase + (size_t)i * 8];
      vl = *(const v4u*)&Alo[gbase + (size_t)i * 8];
    }
    ((v4u*)sB_hi)[i] = vh;
    ((v4u*)sB_lo)[i] = vl;
  }
  for (int i = t0; i < CHUNK; i += tstride)
    aLab[i] = (i < cCount) ? alabels[splitBase + cBase + i] : 0;
}

// ---------------- fused GEMM (bf16x3) + streaming per-row top-50 ----------------
__global__ __launch_bounds__(256)
void knn_gemm_topk(const unsigned short* __restrict__ Fhi,
                   const unsigned short* __restrict__ Flo,
                   const unsigned short* __restrict__ Ahi,
                   const unsigned short* __restrict__ Alo,
                   const int* __restrict__ labels,
                   const int* __restrict__ alabels,
                   float* __restrict__ lists) {
  __shared__ __align__(16) unsigned short sA_hi[BTILE * D];   // 16 KB
  __shared__ __align__(16) unsigned short sA_lo[BTILE * D];   // 16 KB
  __shared__ __align__(16) unsigned short sB_hi[CHUNK * D];   // 64 KB
  __shared__ __align__(16) unsigned short sB_lo[CHUNK * D];   // 64 KB
  __shared__ float cand[BTILE * CHUNK];                       // 16 KB
  __shared__ float topk[BTILE * KTOP];                        // 6.4 KB
  __shared__ float thresh[BTILE];
  __shared__ float rowMin[BTILE];
  __shared__ int   minPos[BTILE];
  __shared__ int   cnt[BTILE];
  __shared__ int   candCnt[BTILE];
  __shared__ int   sLab[BTILE];
  __shared__ int   aLab[CHUNK];

  const int tid       = threadIdx.x;
  const int bt        = blockIdx.x / NSPLIT;
  const int sp        = blockIdx.x % NSPLIT;
  const int rowBase   = bt * BTILE;
  const int splitBase = sp * SPLIT_LEN;
  const int nChunks   = (SPLIT_LEN + CHUNK - 1) / CHUNK;

  // Prologue: load A tile (bf16 hi/lo), labels, init top-k state, stage chunk 0.
  {
    const v4u* gh = (const v4u*)(Fhi + (size_t)rowBase * D);
    const v4u* gl = (const v4u*)(Flo + (size_t)rowBase * D);
    for (int i = tid; i < BTILE * (D / 8); i += 256) {
      ((v4u*)sA_hi)[i] = gh[i];
      ((v4u*)sA_lo)[i] = gl[i];
    }
  }
  if (tid < BTILE) {
    sLab[tid]    = labels[rowBase + tid];
    thresh[tid]  = -3.0e38f;
    rowMin[tid]  = -3.0e38f;
    minPos[tid]  = 0;
    cnt[tid]     = 0;
    candCnt[tid] = 0;
  }
  for (int i = tid; i < BTILE * KTOP; i += 256)
    topk[i] = __uint_as_float(0xFF800000u);  // -inf, match bit 0
  stage_chunk(Ahi, Alo, alabels, sB_hi, sB_lo, aLab, splitBase, 0, tid, 256);
  __syncthreads();

  // Tile mapping: wave w owns M-tile (w&1) and N-tiles {2*(w>>1), 2*(w>>1)+1}.
  const int wave = tid >> 5;   // wave32: 8 waves
  const int lane = tid & 31;
  const int n16  = lane & 15;  // col within 16-wide B tile / row within A M-tile
  const int h    = lane >> 4;  // lane half selects contiguous K half 16h..16h+15 (A and B alike)
  const int m    = wave & 1;
  const int n0   = (wave >> 1) * 2;

  // Hoist this wave's A fragments (hi+lo, all 8 K-slices) into registers:
  // A is invariant for the whole kernel, so these 128 VGPRs kill 1/3 of the
  // inner-loop LDS traffic (the CU's shared LDS path is the gating pipe).
  FragU aH[8], aL[8];
#pragma unroll
  for (int s = 0; s < 8; s++) {
    const int aoff = (m * 16 + n16) * D + s * 32 + h * 16;
    aH[s].q[0] = *(const v4u*)&sA_hi[aoff];
    aH[s].q[1] = *(const v4u*)&sA_hi[aoff + 8];
    aL[s].q[0] = *(const v4u*)&sA_lo[aoff];
    aL[s].q[1] = *(const v4u*)&sA_lo[aoff + 8];
  }

  for (int c = 0; c < nChunks; c++) {
    const int cBase  = c * CHUNK;
    const int rem    = SPLIT_LEN - cBase;
    const int cCount = (rem < CHUNK) ? rem : CHUNK;

    // ---- compute: two 16x16 C tiles per wave (one M-tile, two N-tiles) ----
    v8f acc0, acc1;
    {
      v8f z = {0.f, 0.f, 0.f, 0.f, 0.f, 0.f, 0.f, 0.f};
      acc0 = z; acc1 = z;
    }
#pragma unroll
    for (int s = 0; s < 8; s++) {
      const int kb = s * 32;
      FragU bh0, bl0, bh1, bl1;  // B 32x16: lane col=n16, halves hold K kb+h*16..+15
      const int boff0 = (n0 * 16 + n16) * D + kb + h * 16;
      const int boff1 = boff0 + 16 * D;
      bh0.q[0] = *(const v4u*)&sB_hi[boff0];
      bh0.q[1] = *(const v4u*)&sB_hi[boff0 + 8];
      bl0.q[0] = *(const v4u*)&sB_lo[boff0];
      bl0.q[1] = *(const v4u*)&sB_lo[boff0 + 8];
      bh1.q[0] = *(const v4u*)&sB_hi[boff1];
      bh1.q[1] = *(const v4u*)&sB_hi[boff1 + 8];
      bl1.q[0] = *(const v4u*)&sB_lo[boff1];
      bl1.q[1] = *(const v4u*)&sB_lo[boff1 + 8];
      // bf16x3: hi*hi + lo*hi + hi*lo  (~fp32-faithful ranking)
      acc0 = wmma_bf16(aH[s].v, bh0.v, acc0);
      acc0 = wmma_bf16(aL[s].v, bh0.v, acc0);
      acc0 = wmma_bf16(aH[s].v, bl0.v, acc0);
      acc1 = wmma_bf16(aH[s].v, bh1.v, acc1);
      acc1 = wmma_bf16(aL[s].v, bh1.v, acc1);
      acc1 = wmma_bf16(aH[s].v, bl1.v, acc1);
    }

    // ---- append candidates passing row threshold (match bit in score LSB) ----
    const int acol0 = n0 * 16 + n16;
    const int acol1 = acol0 + 16;
    if (acol0 < cCount) {
#pragma unroll
      for (int r = 0; r < 8; r++) {
        const int arow = m * 16 + r + h * 8;  // C layout: M = r + 8*(lane/16)
        float sc = acc0[r];
        int mt = (sLab[arow] == aLab[acol0]) ? 1 : 0;
        float v = __uint_as_float((__float_as_uint(sc) & ~1u) | (unsigned)mt);
        if (v > thresh[arow]) {
          int p = atomicAdd(&candCnt[arow], 1);   // ds_add_rtn_u32
          cand[arow * CHUNK + p] = v;
        }
      }
    }
    if (acol1 < cCount) {
#pragma unroll
      for (int r = 0; r < 8; r++) {
        const int arow = m * 16 + r + h * 8;
        float sc = acc1[r];
        int mt = (sLab[arow] == aLab[acol1]) ? 1 : 0;
        float v = __uint_as_float((__float_as_uint(sc) & ~1u) | (unsigned)mt);
        if (v > thresh[arow]) {
          int p = atomicAdd(&candCnt[arow], 1);
          cand[arow * CHUNK + p] = v;
        }
      }
    }
    __syncthreads();

    // ---- wave 0 merges candidates; waves 1-7 stage next chunk (overlapped) ----
    if (tid < BTILE) {
      const int r = tid;
      int   nc  = candCnt[r];
      int   cc2 = cnt[r];
      float mn  = rowMin[r];
      int   mp  = minPos[r];
      float* b  = &topk[r * KTOP];
      for (int i = 0; i < nc; i++) {
        float v = cand[r * CHUNK + i];
        if (cc2 < KTOP) {
          b[cc2++] = v;
          if (cc2 == KTOP) {
            mn = b[0]; mp = 0;
            for (int j = 1; j < KTOP; j++) { float t = b[j]; if (t < mn) { mn = t; mp = j; } }
          }
        } else if (v > mn) {
          b[mp] = v;
          mn = b[0]; mp = 0;
          for (int j = 1; j < KTOP; j++) { float t = b[j]; if (t < mn) { mn = t; mp = j; } }
        }
      }
      cnt[r] = cc2; rowMin[r] = mn; minPos[r] = mp;
      thresh[r] = (cc2 >= KTOP) ? mn : -3.0e38f;
      candCnt[r] = 0;
    } else if (c + 1 < nChunks) {
      stage_chunk(Ahi, Alo, alabels, sB_hi, sB_lo, aLab, splitBase, c + 1,
                  tid - 32, 224);
    }
    __syncthreads();
  }

  // Epilogue: write this split's per-row top-50 list (match bit in LSB).
  for (int i = tid; i < BTILE * KTOP; i += 256) {
    int r = i / KTOP, j = i % KTOP;
    lists[((size_t)(rowBase + r) * NSPLIT + sp) * KTOP + j] = topk[r * KTOP + j];
  }
}

// ---------------- merge NSPLIT partial top-50 lists -> exact top-50 ----------------
__global__ __launch_bounds__(256)
void merge_topk(const float* __restrict__ lists, float* __restrict__ blockSums) {
  __shared__ float buf[256 * KTOP];   // 51.2 KB
  __shared__ float red[256];
  const int r = blockIdx.x * 256 + threadIdx.x;
  const float* src = lists + (size_t)r * NSPLIT * KTOP;
  float* b = &buf[threadIdx.x * KTOP];
  int cn = 0, mp = 0;
  float mn = 0.f;
  for (int i = 0; i < NSPLIT * KTOP; i++) {
    float v = src[i];
    if (cn < KTOP) {
      b[cn++] = v;
      if (cn == KTOP) {
        mn = b[0]; mp = 0;
        for (int j = 1; j < KTOP; j++) { float t = b[j]; if (t < mn) { mn = t; mp = j; } }
      }
    } else if (v > mn) {
      b[mp] = v;
      mn = b[0]; mp = 0;
      for (int j = 1; j < KTOP; j++) { float t = b[j]; if (t < mn) { mn = t; mp = j; } }
    }
  }
  int ms = 0;
  for (int j = 0; j < KTOP; j++) ms += (int)(__float_as_uint(b[j]) & 1u);
  red[threadIdx.x] = (float)ms / (float)KTOP;
  __syncthreads();
  for (int s = 128; s > 0; s >>= 1) {
    if (threadIdx.x < s) red[threadIdx.x] += red[threadIdx.x + s];
    __syncthreads();
  }
  if (threadIdx.x == 0) blockSums[blockIdx.x] = red[0];
}

__global__ void finalize(const float* __restrict__ blockSums, float* __restrict__ out) {
  if (threadIdx.x == 0 && blockIdx.x == 0) {
    float s = 0.f;
    for (int i = 0; i < NB / 256; i++) s += blockSums[i];
    out[0] = -(s / (float)NB);
  }
}

// ---------------- host-side launch ----------------
extern "C" void kernel_launch(void* const* d_in, const int* in_sizes, int n_in,
                              void* d_out, int out_size, void* d_ws, size_t ws_size,
                              hipStream_t stream) {
  const float* features = (const float*)d_in[0];  // [2048,256] f32
  const float* anchors  = (const float*)d_in[1];  // [50000,256] f32
  const int*   labels   = (const int*)d_in[2];    // [2048] int
  const int*   alabels  = (const int*)d_in[3];    // [50000] int

  unsigned char* ws = (unsigned char*)d_ws;
  const size_t OFF_FHI = 0;
  const size_t OFF_FLO = OFF_FHI + (size_t)NB * D * 2;   //  1 MB
  const size_t OFF_AHI = OFF_FLO + (size_t)NB * D * 2;   //  2 MB
  const size_t OFF_ALO = OFF_AHI + (size_t)NA * D * 2;   // 27.7 MB
  const size_t OFF_LST = OFF_ALO + (size_t)NA * D * 2;   // 53.3 MB
  const size_t OFF_BSM = OFF_LST + (size_t)NB * NSPLIT * KTOP * 4;

  unsigned short* Fhi = (unsigned short*)(ws + OFF_FHI);
  unsigned short* Flo = (unsigned short*)(ws + OFF_FLO);
  unsigned short* Ahi = (unsigned short*)(ws + OFF_AHI);
  unsigned short* Alo = (unsigned short*)(ws + OFF_ALO);
  float* lists        = (float*)(ws + OFF_LST);
  float* blockSums    = (float*)(ws + OFF_BSM);

  split_convert<<<256, 256, 0, stream>>>(features, Fhi, Flo, NB * D / 4);
  split_convert<<<1024, 256, 0, stream>>>(anchors, Ahi, Alo, NA * D / 4);
  knn_gemm_topk<<<(NB / BTILE) * NSPLIT, 256, 0, stream>>>(
      Fhi, Flo, Ahi, Alo, labels, alabels, lists);
  merge_topk<<<NB / 256, 256, 0, stream>>>(lists, blockSums);
  finalize<<<1, 32, 0, stream>>>(blockSums, (float*)d_out);
}
